// SpatialTemporalGraphConv_85332410237414
// MI455X (gfx1250) — matrix operand order repaired and checked by reading, not compiled, not run on hardware
//
#include <hip/hip_runtime.h>
#include <hip/hip_bf16.h>

// ---------------------------------------------------------------------------
// Spatial-Temporal Graph Conv for MI455X (gfx1250, wave32, WMMA bf16)
// B=16 N=12 C=32 L=5000 D=64 K=9. Dominant cost: temporal conv (70.8 GFLOP)
// -> bf16 WMMA GEMM (v_wmma_f32_16x16x32_bf16) with f32 accumulate.
// conv_b / res_b cancel through training-mode BN and are dropped; b_gcn kept
// (edge effects via conv zero-padding).
// ---------------------------------------------------------------------------

typedef __attribute__((ext_vector_type(16))) __bf16 v16bf;
typedef __attribute__((ext_vector_type(8)))  float  v8f;

#define L_    5000
#define D_    64
#define C_    32
#define NN_   12
#define BB_   16
#define BN_   192          // B*N
#define KK_   9
#define PADK_ 4
#define KTOT_ 576          // 9*64, reordered K = kc*64 + e
#define SXS_  72           // LDS stride (bf16) for conv tile: 144B, 16B-aligned
#define SYS_  40           // LDS stride (bf16) for gcn tile:   80B, 16B-aligned
#define TT_   128          // time-tile per block (8 waves x 16)

struct B32x8 { float4 a, b; };  // 32-byte carrier for a v16bf fragment

// 16 contiguous bf16 (used for B fragments from LDS)
static __device__ __forceinline__ v16bf ld_frag16(const __bf16* p) {
  B32x8 t;
  t.a = *reinterpret_cast<const float4*>(p);
  t.b = *reinterpret_cast<const float4*>(p + 8);
  return __builtin_bit_cast(v16bf, t);
}

// A fragment: lane (m=col, half) holds K run [c*32+16h .. +15]
static __device__ __forceinline__ v16bf ld_afrag(const __bf16* row, int c, int half) {
  const __bf16* p = row + c * 32 + half * 16;
  B32x8 t;
  t.a = *reinterpret_cast<const float4*>(p);
  t.b = *reinterpret_cast<const float4*>(p + 8);
  return __builtin_bit_cast(v16bf, t);
}

// ---------------------------------------------------------------------------
// Weight repack to bf16: W2[m][kc*64+e] = conv_w[m][e][kc]; WTg[d][c] = w_gcn[c][d];
// WTr[d][c] = res_w[d][c][0]
// ---------------------------------------------------------------------------
__global__ __launch_bounds__(256)
void stgc_prep(const float* __restrict__ w_gcn, const float* __restrict__ conv_w,
               const float* __restrict__ res_w,
               __bf16* __restrict__ W2, __bf16* __restrict__ WTg, __bf16* __restrict__ WTr) {
  const int total = D_ * KTOT_ + D_ * C_ + D_ * C_;
  for (int idx = blockIdx.x * 256 + threadIdx.x; idx < total; idx += gridDim.x * 256) {
    if (idx < D_ * KTOT_) {
      int m = idx / KTOT_, km = idx - m * KTOT_;
      int kc = km >> 6, e = km & 63;
      W2[idx] = (__bf16)conv_w[(m * D_ + e) * KK_ + kc];
    } else if (idx < D_ * KTOT_ + D_ * C_) {
      int j = idx - D_ * KTOT_;
      int dd = j >> 5, cc = j & 31;
      WTg[j] = (__bf16)w_gcn[cc * D_ + dd];
    } else {
      int j = idx - D_ * KTOT_ - D_ * C_;
      WTr[j] = (__bf16)res_w[j];
    }
  }
}

// ---------------------------------------------------------------------------
// GCN / residual GEMM: out[bn][d][t] = sum_c WT[d][c] * y[c][t] (+ bias[d])
//   USE_ADJ=1: y[c][t] = sum_m adj[b,n,m] * x[b,m,c,t]   (GCN path)
//   USE_ADJ=0: y[c][t] = x[b,n,c,t]                       (residual path)
// One 128-t tile per block; 8 waves x (64x16) WMMA tiles, K=C=32 = one chunk.
// ---------------------------------------------------------------------------
template <bool USE_ADJ, bool HAS_BIAS>
__global__ __launch_bounds__(256)
void stgc_gemm32(const float* __restrict__ x, const float* __restrict__ adj,
                 const __bf16* __restrict__ WT, const float* __restrict__ bias,
                 float* __restrict__ out) {
  __shared__ __align__(16) __bf16 sY[TT_ * SYS_];
  const int bn = blockIdx.x;
  const int b = bn / NN_, n = bn - b * NN_;
  const int t0 = blockIdx.y * TT_;
  const int tid = threadIdx.x;

  float arow[NN_] = {};
  if constexpr (USE_ADJ) {
    #pragma unroll
    for (int m = 0; m < NN_; ++m) arow[m] = adj[(b * NN_ + n) * NN_ + m];
  }

  for (int idx = tid; idx < C_ * TT_; idx += 256) {
    int cch = idx >> 7;        // /128
    int tl  = idx & 127;
    int t   = t0 + tl;
    float v = 0.0f;
    if (t < L_) {
      if constexpr (USE_ADJ) {
        #pragma unroll
        for (int m = 0; m < NN_; ++m)
          v += arow[m] * x[((b * NN_ + m) * C_ + cch) * L_ + t];
      } else {
        v = x[((b * NN_ + n) * C_ + cch) * L_ + t];
      }
    }
    sY[tl * SYS_ + cch] = (__bf16)v;
  }
  __syncthreads();

  const int wave = tid >> 5, lane = tid & 31;
  const int half = lane >> 4, col = lane & 15;
  const int tw = wave * 16;

  v8f acc[4];
  #pragma unroll
  for (int mt = 0; mt < 4; ++mt)
    #pragma unroll
    for (int r = 0; r < 8; ++r) acc[mt][r] = 0.0f;

  v16bf bf = ld_frag16(&sY[(tw + col) * SYS_ + half * 16]);
  #pragma unroll
  for (int mt = 0; mt < 4; ++mt) {
    v16bf af = ld_afrag(WT + (mt * 16 + col) * C_, 0, half);
    acc[mt] = __builtin_amdgcn_wmma_f32_16x16x32_bf16(
        false, af, false, bf, (short)0, acc[mt], false, false);
  }

  // Hoisted bias: d = mt*16 + half*8 + r -> 8 contiguous floats per M-tile.
  float bsv[32];
  if constexpr (HAS_BIAS) {
    #pragma unroll
    for (int mt = 0; mt < 4; ++mt) {
      const float4* bp = reinterpret_cast<const float4*>(&bias[mt * 16 + half * 8]);
      float4 q0 = bp[0], q1 = bp[1];
      bsv[mt * 8 + 0] = q0.x; bsv[mt * 8 + 1] = q0.y;
      bsv[mt * 8 + 2] = q0.z; bsv[mt * 8 + 3] = q0.w;
      bsv[mt * 8 + 4] = q1.x; bsv[mt * 8 + 5] = q1.y;
      bsv[mt * 8 + 6] = q1.z; bsv[mt * 8 + 7] = q1.w;
    }
  }

  const int t = t0 + tw + col;
  if (t < L_) {
    #pragma unroll
    for (int mt = 0; mt < 4; ++mt) {
      float* o = &out[(bn * D_ + mt * 16 + half * 8) * L_ + t];
      #pragma unroll
      for (int r = 0; r < 8; ++r) {
        float v = acc[mt][r];
        if constexpr (HAS_BIAS) v += bsv[mt * 8 + r];
        o[r * (long)L_] = v;
      }
    }
  }
}

// ---------------------------------------------------------------------------
// Temporal conv as GEMM: h[d][t] = sum_{K=kc*64+e} W2[d][K] * xs[e][t+kc-4]
// Stage 64ch x 136t (halo incl., zero-padded) in LDS bf16; 18 K-chunks of 32.
// ---------------------------------------------------------------------------
__global__ __launch_bounds__(256)
void stgc_conv(const float* __restrict__ xs, const __bf16* __restrict__ W2,
               float* __restrict__ h) {
  __shared__ __align__(16) __bf16 sX[136 * SXS_];
  const int bn = blockIdx.x;
  const int t0 = blockIdx.y * TT_;
  const int tid = threadIdx.x;

  for (int idx = tid; idx < D_ * 136; idx += 256) {
    int e  = idx / 136;
    int tl = idx - e * 136;
    int t  = t0 - PADK_ + tl;
    float v = (t >= 0 && t < L_) ? xs[(bn * D_ + e) * L_ + t] : 0.0f;
    sX[tl * SXS_ + e] = (__bf16)v;
  }
  __syncthreads();

  const int wave = tid >> 5, lane = tid & 31;
  const int half = lane >> 4, col = lane & 15;
  const int tw = wave * 16;

  v8f acc[4];
  #pragma unroll
  for (int mt = 0; mt < 4; ++mt)
    #pragma unroll
    for (int r = 0; r < 8; ++r) acc[mt][r] = 0.0f;

  for (int c = 0; c < 18; ++c) {
    int kc = c >> 1;
    int eb = (c & 1) << 5;
    // B fragment: K run = eb + half*16 + [0..15] at time (t0+tw+col)+kc-4
    v16bf bf = ld_frag16(&sX[(tw + col + kc) * SXS_ + eb + half * 16]);
    #pragma unroll
    for (int mt = 0; mt < 4; ++mt) {
      v16bf af = ld_afrag(W2 + (mt * 16 + col) * KTOT_, c, half);
      acc[mt] = __builtin_amdgcn_wmma_f32_16x16x32_bf16(
          false, af, false, bf, (short)0, acc[mt], false, false);
    }
  }

  const int t = t0 + tw + col;
  if (t < L_) {
    #pragma unroll
    for (int mt = 0; mt < 4; ++mt) {
      float* o = &h[(bn * D_ + mt * 16 + half * 8) * L_ + t];
      #pragma unroll
      for (int r = 0; r < 8; ++r) o[r * (long)L_] = acc[mt][r];
    }
  }
}

// ---------------------------------------------------------------------------
// BN stats: per (n,d) over (b,t) — 768 blocks, two-moment block reduction
// ---------------------------------------------------------------------------
__global__ __launch_bounds__(256)
void stgc_stats_nd(const float* __restrict__ h, float* __restrict__ mean,
                   float* __restrict__ istd) {
  const int nd = blockIdx.x;
  const int n = nd >> 6, d = nd & 63;
  float s = 0.0f, sq = 0.0f;
  for (int b = 0; b < BB_; ++b) {
    const float* p = &h[((b * NN_ + n) * D_ + d) * (long)L_];
    for (int t = threadIdx.x; t < L_; t += 256) {
      float v = p[t];
      s += v; sq += v * v;
    }
  }
  __shared__ float rs[256], rq[256];
  rs[threadIdx.x] = s; rq[threadIdx.x] = sq;
  __syncthreads();
  for (int off = 128; off > 0; off >>= 1) {
    if (threadIdx.x < off) {
      rs[threadIdx.x] += rs[threadIdx.x + off];
      rq[threadIdx.x] += rq[threadIdx.x + off];
    }
    __syncthreads();
  }
  if (threadIdx.x == 0) {
    const float cnt = (float)(BB_ * L_);
    float m = rs[0] / cnt;
    float var = rq[0] / cnt - m * m;
    mean[nd] = m;
    istd[nd] = rsqrtf(var + 1e-5f);
  }
}

// BN stats: per d over (b,n,t) — 64 blocks
__global__ __launch_bounds__(256)
void stgc_stats_d(const float* __restrict__ r, float* __restrict__ mean,
                  float* __restrict__ istd) {
  const int d = blockIdx.x;
  float s = 0.0f, sq = 0.0f;
  for (int bn = 0; bn < BN_; ++bn) {
    const float* p = &r[(bn * D_ + d) * (long)L_];
    for (int t = threadIdx.x; t < L_; t += 256) {
      float v = p[t];
      s += v; sq += v * v;
    }
  }
  __shared__ float rs[256], rq[256];
  rs[threadIdx.x] = s; rq[threadIdx.x] = sq;
  __syncthreads();
  for (int off = 128; off > 0; off >>= 1) {
    if (threadIdx.x < off) {
      rs[threadIdx.x] += rs[threadIdx.x + off];
      rq[threadIdx.x] += rq[threadIdx.x + off];
    }
    __syncthreads();
  }
  if (threadIdx.x == 0) {
    const float cnt = (float)(BN_ * L_);
    float m = rs[0] / cnt;
    float var = rq[0] / cnt - m * m;
    mean[d] = m;
    istd[d] = rsqrtf(var + 1e-5f);
  }
}

// ---------------------------------------------------------------------------
// Finalize: out = relu( relu(bn1(h)) + bn2(r) )
// ---------------------------------------------------------------------------
__global__ __launch_bounds__(256)
void stgc_final(const float* __restrict__ h, const float* __restrict__ r,
                const float* __restrict__ m1, const float* __restrict__ is1,
                const float* __restrict__ g1, const float* __restrict__ b1,
                const float* __restrict__ m2, const float* __restrict__ is2,
                const float* __restrict__ g2, const float* __restrict__ b2,
                float* __restrict__ out) {
  const int total = BN_ * D_ * L_;   // 61,440,000
  for (int idx = blockIdx.x * 256 + threadIdx.x; idx < total;
       idx += gridDim.x * 256) {
    int rest = idx / L_;
    int d  = rest & 63;
    int bn = rest >> 6;
    int nd = (bn % NN_) * 64 + d;
    float hv = (h[idx] - m1[nd]) * is1[nd] * g1[d] + b1[d];
    hv = fmaxf(hv, 0.0f);
    float rv = (r[idx] - m2[d]) * is2[d] * g2[d] + b2[d];
    out[idx] = fmaxf(hv + rv, 0.0f);
  }
}

// ---------------------------------------------------------------------------
extern "C" void kernel_launch(void* const* d_in, const int* in_sizes, int n_in,
                              void* d_out, int out_size, void* d_ws, size_t ws_size,
                              hipStream_t stream) {
  (void)in_sizes; (void)n_in; (void)out_size; (void)ws_size;
  const float* x      = (const float*)d_in[0];
  const float* adj    = (const float*)d_in[1];
  const float* w_gcn  = (const float*)d_in[2];
  const float* b_gcn  = (const float*)d_in[3];
  const float* conv_w = (const float*)d_in[4];
  // d_in[5] conv_b: cancels through train-mode BN (per-(n,d) mean over b,t)
  const float* bn1g   = (const float*)d_in[6];
  const float* bn1b   = (const float*)d_in[7];
  const float* res_w  = (const float*)d_in[8];
  // d_in[9] res_b: cancels through train-mode BN (per-d mean over b,n,t)
  const float* bn2g   = (const float*)d_in[10];
  const float* bn2b   = (const float*)d_in[11];

  const size_t NE = (size_t)BN_ * D_ * L_;     // 61.44M floats
  float*  xs = (float*)d_ws;                   // [0, NE)
  float*  h  = xs + NE;                        // [NE, 2NE)
  float*  rr = xs;                             // reuse xs buffer after conv
  __bf16* wb  = (__bf16*)(h + NE);
  __bf16* W2  = wb;                            // 36864
  __bf16* WTg = wb + D_ * KTOT_;               // 2048
  __bf16* WTr = WTg + D_ * C_;                 // 2048
  float*  st  = (float*)(WTr + D_ * C_ + 64);  // stats, 4-byte aligned
  float*  m1  = st;
  float*  is1 = m1 + NN_ * D_;
  float*  m2  = is1 + NN_ * D_;
  float*  is2 = m2 + D_;

  dim3 blk(256);
  dim3 gtile(BN_, (L_ + TT_ - 1) / TT_);       // 192 x 40

  stgc_prep<<<dim3(160), blk, 0, stream>>>(w_gcn, conv_w, res_w, W2, WTg, WTr);
  stgc_gemm32<true, true><<<gtile, blk, 0, stream>>>(x, adj, WTg, b_gcn, xs);
  stgc_conv<<<gtile, blk, 0, stream>>>(xs, W2, h);
  stgc_stats_nd<<<dim3(NN_ * D_), blk, 0, stream>>>(h, m1, is1);
  stgc_gemm32<false, false><<<gtile, blk, 0, stream>>>(x, adj, WTr, nullptr, rr);
  stgc_stats_d<<<dim3(D_), blk, 0, stream>>>(rr, m2, is2);
  stgc_final<<<dim3(2048), blk, 0, stream>>>(h, rr, m1, is1, bn1g, bn1b,
                                             m2, is2, bn2g, bn2b, (float*)d_out);
}
